// Node_align_Node_loss_21680994910651
// MI455X (gfx1250) — matrix operand order repaired and checked by reading, not compiled, not run on hardware
//
#include <hip/hip_runtime.h>
#include <hip/hip_bf16.h>
#include <stdint.h>

// ---------------------------------------------------------------------------
// GMN forward for MI455X (gfx1250): all GEMM work via v_wmma_f32_16x16x32_bf16
// (fp32 accumulate). wave32; A tiles staged row-major in LDS, read in contiguous-k order,
// B fragments pre-packed in global so each lane does 2x b128 loads.
// Dual accumulators per output tile break the WMMA->WMMA RAW chain; weight
// fragments are prefetched (global_prefetch_b8) two K-steps ahead.
// ---------------------------------------------------------------------------

typedef __attribute__((ext_vector_type(16))) __bf16 bf16x16;
typedef __attribute__((ext_vector_type(8)))  float  f32x8;

#define N_TOTAL  26624      // B*(NQ+NC)
#define E_TOTAL  106496     // N_TOTAL*DEG
#define BPAIR    256
#define DIM      128
#define ESTATE   32
#define MSGW     256
#define AW       288        // 2*DIM + ESTATE
#define UPW      384        // MSGW + DIM
#define INV_TEMP 10.0f
#define NQ_      48
#define NC_      56

__device__ __forceinline__ f32x8 wmma_bf(bf16x16 a, bf16x16 b, f32x8 c) {
  return __builtin_amdgcn_wmma_f32_16x16x32_bf16(false, a, false, b, (short)0, c,
                                                 false, false);
}

// A-matrix 16x32 bf16 fragment from LDS (row-major [16][stride]).
// Layout (contiguous-k order, same as packed B): lanes 0-15/16-31 both cover M=0..15; VGPR v holds
// K = (lane>=16?16:0) + 2*v .. +1
__device__ __forceinline__ bf16x16 load_a_frag(const __bf16* lds, int stride, int lane) {
  bf16x16 a;
  const __bf16* row = lds + (lane & 15) * stride;
  int kh = ((lane >> 4) & 1) << 4;
#pragma unroll
  for (int v = 0; v < 8; ++v) {
    int k = kh + (v << 1);
    union { unsigned u; __bf16 b[2]; } cv;
    cv.u = *(const unsigned*)(row + k);
    a[2 * v] = cv.b[0];
    a[2 * v + 1] = cv.b[1];
  }
  return a;
}

// B-matrix 32x16 fragment from pre-packed global: lane l holds 16 contiguous bf16.
__device__ __forceinline__ bf16x16 load_b_frag(const __bf16* pk, int frag, int lane) {
  union { uint4 q[2]; bf16x16 v; } u;
  const uint4* s = (const uint4*)(pk + ((size_t)frag * 32 + lane) * 16);
  u.q[0] = s[0];
  u.q[1] = s[1];
  return u.v;
}

__device__ __forceinline__ void prefetch_b_frag(const __bf16* pk, int frag, int lane) {
  __builtin_prefetch(pk + ((size_t)frag * 32 + lane) * 16, 0, 1);
}

// B fragment assembled from an LDS row-major matrix where B[k][n] = M[n][k]
// (i.e. computing X @ M^T): lane reads 16 contiguous bf16 from row n.
__device__ __forceinline__ bf16x16 load_b_frag_lds(const __bf16* lds, int stride,
                                                   int n0, int k0, int lane) {
  bf16x16 b;
  const __bf16* src = lds + (size_t)(n0 + (lane & 15)) * stride + k0 +
                      (((lane >> 4) & 1) << 4);
#pragma unroll
  for (int i = 0; i < 8; ++i) {
    union { unsigned u; __bf16 b2[2]; } cv;
    cv.u = *(const unsigned*)(src + 2 * i);
    b[2 * i] = cv.b2[0];
    b[2 * i + 1] = cv.b2[1];
  }
  return b;
}

// ---------------------------------------------------------------------------
// Weight packing: fp32 W[K][Nw] -> bf16 fragments [(kt*ntiles+nt)*32+lane][16]
// lane l: n = nt*16 + (l&15), k = kt*32 + (l>=16?16:0) + i.  Zero-pad k >= K.
// ---------------------------------------------------------------------------
__global__ void pack_weights_kernel(const float* __restrict__ W, __bf16* __restrict__ out,
                                    int K, int Nw, int ntiles, int total_threads) {
  int idx = blockIdx.x * blockDim.x + threadIdx.x;
  if (idx >= total_threads) return;
  int lane = idx & 31;
  int frag = idx >> 5;
  int nt = frag % ntiles;
  int kt = frag / ntiles;
  int n = nt * 16 + (lane & 15);
  int kbase = kt * 32 + ((lane >> 4) & 1) * 16;
  __bf16* o = out + (size_t)idx * 16;
#pragma unroll
  for (int i = 0; i < 16; ++i) {
    int k = kbase + i;
    float v = (k < K) ? W[(size_t)k * Nw + n] : 0.0f;
    o[i] = (__bf16)v;
  }
}

// ---------------------------------------------------------------------------
// Node encoder: h[N,128] = nf[N,32] @ Wn + bn   (one WMMA per 16x16 tile, K=32)
// ---------------------------------------------------------------------------
__global__ __launch_bounds__(256) void encode_nodes_kernel(
    const float* __restrict__ nf, const __bf16* __restrict__ wnp,
    const float* __restrict__ bn, float* __restrict__ h, __bf16* __restrict__ h_bf) {
  __shared__ __bf16 sA[8][16 * 32];
  int wave = threadIdx.x >> 5, lane = threadIdx.x & 31;
  int row0 = (blockIdx.x * 8 + wave) * 16;
  if (row0 >= N_TOTAL) return;
  __bf16* A = sA[wave];
  {
    int m = lane >> 1, c0 = (lane & 1) * 16;
    const float* src = nf + (size_t)(row0 + m) * 32 + c0;
#pragma unroll
    for (int j = 0; j < 16; ++j) A[m * 32 + c0 + j] = (__bf16)src[j];
  }
  asm volatile("s_wait_dscnt 0" ::: "memory");
  bf16x16 a = load_a_frag(A, 32, lane);
#pragma unroll 1
  for (int nt = 0; nt < 8; ++nt) {
    f32x8 z = {};
    bf16x16 b = load_b_frag(wnp, nt, lane);
    f32x8 acc = wmma_bf(a, b, z);
    int n = nt * 16 + (lane & 15);
    float bias = bn[n];
    int mb = ((lane >> 4) & 1) * 8;
#pragma unroll
    for (int r = 0; r < 8; ++r) {
      float v = acc[r] + bias;
      size_t o = (size_t)(row0 + mb + r) * DIM + n;
      h[o] = v;
      h_bf[o] = (__bf16)v;
    }
  }
}

// ---------------------------------------------------------------------------
// Edge encoder: e[E,32] = ef[E,16] @ We + be  (K padded 16->32 with zeros)
// ---------------------------------------------------------------------------
__global__ __launch_bounds__(256) void encode_edges_kernel(
    const float* __restrict__ ef, const __bf16* __restrict__ wep,
    const float* __restrict__ be, __bf16* __restrict__ e_bf) {
  __shared__ __bf16 sA[8][16 * 32];
  int wave = threadIdx.x >> 5, lane = threadIdx.x & 31;
  int tile = blockIdx.x * 8 + wave;
  if (tile * 16 >= E_TOTAL) return;
  int row0 = tile * 16;
  __bf16* A = sA[wave];
  {
    int m = lane >> 1, half = lane & 1;
    if (!half) {
      const float* src = ef + (size_t)(row0 + m) * 16;
#pragma unroll
      for (int j = 0; j < 16; ++j) A[m * 32 + j] = (__bf16)src[j];
    } else {
#pragma unroll
      for (int j = 0; j < 16; ++j) A[m * 32 + 16 + j] = (__bf16)0.0f;
    }
  }
  asm volatile("s_wait_dscnt 0" ::: "memory");
  bf16x16 a = load_a_frag(A, 32, lane);
#pragma unroll 1
  for (int nt = 0; nt < 2; ++nt) {
    f32x8 z = {};
    bf16x16 b = load_b_frag(wep, nt, lane);
    f32x8 acc = wmma_bf(a, b, z);
    int n = nt * 16 + (lane & 15);
    float bias = be[n];
    int mb = ((lane >> 4) & 1) * 8;
#pragma unroll
    for (int r = 0; r < 8; ++r)
      e_bf[(size_t)(row0 + mb + r) * ESTATE + n] = (__bf16)(acc[r] + bias);
  }
}

// ---------------------------------------------------------------------------
// Fused message pass (fwd + rev) for one round. One wave per 16-edge tile:
// gather -> [hf|ht|e] (288) -> MLP(288->256 relu ->256) -> atomic scatter.
// Reverse direction just swaps the hf/ht column blocks and params/targets.
// ---------------------------------------------------------------------------
__device__ __forceinline__ int msg_col(int kt, int dir) {
  if (kt < 4) return (dir ? 128 : 0) + kt * 32;
  if (kt < 8) return (dir ? 0 : 128) + (kt - 4) * 32;
  return 256;
}

__global__ __launch_bounds__(128) void msg_pass_kernel(
    const __bf16* __restrict__ h_bf, const __bf16* __restrict__ e_bf,
    const int* __restrict__ from_idx, const int* __restrict__ to_idx,
    const __bf16* __restrict__ wm1p, const float* __restrict__ bm1,
    const __bf16* __restrict__ wm2p, const float* __restrict__ bm2,
    const __bf16* __restrict__ wr1p, const float* __restrict__ br1,
    const __bf16* __restrict__ wr2p, const float* __restrict__ br2,
    float* __restrict__ agg) {
  __shared__ __bf16 sA[4][16 * AW];
  __shared__ __bf16 sH[4][16 * MSGW];
  __shared__ int sIdx[4][32];
  int wave = threadIdx.x >> 5, lane = threadIdx.x & 31;
  int tile = blockIdx.x * 4 + wave;
  if (tile * 16 >= E_TOTAL) return;
  int e0 = tile * 16;
  __bf16* A = sA[wave];
  __bf16* Hd = sH[wave];
  int* idx = sIdx[wave];
  if (lane < 16) {
    idx[lane] = from_idx[e0 + lane];
    idx[16 + lane] = to_idx[e0 + lane];
  }
  asm volatile("s_wait_dscnt 0" ::: "memory");
  // gather A rows: [hf(128) | ht(128) | e(32)] as bf16
  for (int m = 0; m < 16; ++m) {
    const unsigned* hf = (const unsigned*)(h_bf + (size_t)idx[m] * DIM);
    const unsigned* ht = (const unsigned*)(h_bf + (size_t)idx[16 + m] * DIM);
    const unsigned* ep = (const unsigned*)(e_bf + (size_t)(e0 + m) * ESTATE);
    unsigned* dst = (unsigned*)(A + m * AW);
#pragma unroll
    for (int j = 0; j < 2; ++j) {
      dst[lane + j * 32] = hf[lane + j * 32];
      dst[64 + lane + j * 32] = ht[lane + j * 32];
    }
    if (lane < 16) dst[128 + lane] = ep[lane];
  }
  asm volatile("s_wait_dscnt 0" ::: "memory");

#pragma unroll 1
  for (int dir = 0; dir < 2; ++dir) {
    const __bf16* w1pk = dir ? wr1p : wm1p;
    const float* b1k = dir ? br1 : bm1;
    const __bf16* w2pk = dir ? wr2p : wm2p;
    const float* b2k = dir ? br2 : bm2;
    int scat = dir ? 0 : 16;  // fwd -> to_idx, rev -> from_idx
    // layer 1: [16,288] @ [288,256], relu; dual accumulators over K
#pragma unroll 1
    for (int nt = 0; nt < 16; ++nt) {
      f32x8 acc0 = {}, acc1 = {};
#pragma unroll 1
      for (int kt = 0; kt < 8; kt += 2) {
        bf16x16 a0 = load_a_frag(A + msg_col(kt, dir), AW, lane);
        bf16x16 b0 = load_b_frag(w1pk, kt * 16 + nt, lane);
        bf16x16 a1 = load_a_frag(A + msg_col(kt + 1, dir), AW, lane);
        bf16x16 b1 = load_b_frag(w1pk, (kt + 1) * 16 + nt, lane);
        prefetch_b_frag(w1pk, (kt + 2) * 16 + nt, lane);  // kt+2 <= 8 valid
        acc0 = wmma_bf(a0, b0, acc0);
        acc1 = wmma_bf(a1, b1, acc1);
      }
      {  // K tail: edge-state block (kt = 8)
        bf16x16 a = load_a_frag(A + 256, AW, lane);
        bf16x16 b = load_b_frag(w1pk, 8 * 16 + nt, lane);
        acc0 = wmma_bf(a, b, acc0);
      }
      int n = nt * 16 + (lane & 15);
      float bias = b1k[n];
      int mb = ((lane >> 4) & 1) * 8;
#pragma unroll
      for (int r = 0; r < 8; ++r) {
        float v = acc0[r] + acc1[r] + bias;
        v = v > 0.0f ? v : 0.0f;
        Hd[(mb + r) * MSGW + n] = (__bf16)v;
      }
    }
    // layer 2: [16,256] @ [256,256], atomic scatter into agg
#pragma unroll 1
    for (int nt = 0; nt < 16; ++nt) {
      f32x8 acc0 = {}, acc1 = {};
#pragma unroll 1
      for (int kt = 0; kt < 8; kt += 2) {
        bf16x16 a0 = load_a_frag(Hd + kt * 32, MSGW, lane);
        bf16x16 b0 = load_b_frag(w2pk, kt * 16 + nt, lane);
        bf16x16 a1 = load_a_frag(Hd + (kt + 1) * 32, MSGW, lane);
        bf16x16 b1 = load_b_frag(w2pk, (kt + 1) * 16 + nt, lane);
        if (kt < 6) prefetch_b_frag(w2pk, (kt + 2) * 16 + nt, lane);
        acc0 = wmma_bf(a0, b0, acc0);
        acc1 = wmma_bf(a1, b1, acc1);
      }
      int n = nt * 16 + (lane & 15);
      float bias = b2k[n];
      int mb = ((lane >> 4) & 1) * 8;
#pragma unroll
      for (int r = 0; r < 8; ++r) {
        int dstrow = idx[scat + mb + r];
        atomicAdd(&agg[(size_t)dstrow * MSGW + n], acc0[r] + acc1[r] + bias);
      }
    }
  }
}

// ---------------------------------------------------------------------------
// Residual node update: h += [agg(256) | h(128)] @ wu + bu   (row-local, safe)
// ---------------------------------------------------------------------------
__global__ __launch_bounds__(128) void node_update_kernel(
    const float* __restrict__ agg, const __bf16* __restrict__ wup,
    const float* __restrict__ bu, float* __restrict__ h, __bf16* __restrict__ h_bf) {
  __shared__ __bf16 sA[4][16 * UPW];
  int wave = threadIdx.x >> 5, lane = threadIdx.x & 31;
  int row0 = (blockIdx.x * 4 + wave) * 16;
  if (row0 >= N_TOTAL) return;
  __bf16* A = sA[wave];
  for (int m = 0; m < 16; ++m) {
    const float4* ag4 = (const float4*)(agg + (size_t)(row0 + m) * MSGW);
#pragma unroll
    for (int j = 0; j < 2; ++j) {
      float4 v4 = ag4[lane + j * 32];
      __bf16* d = A + m * UPW + (lane + j * 32) * 4;
      d[0] = (__bf16)v4.x;
      d[1] = (__bf16)v4.y;
      d[2] = (__bf16)v4.z;
      d[3] = (__bf16)v4.w;
    }
    const unsigned* hb = (const unsigned*)(h_bf + (size_t)(row0 + m) * DIM);
    unsigned* dst = (unsigned*)(A + m * UPW + 256);
#pragma unroll
    for (int j = 0; j < 2; ++j) dst[lane + j * 32] = hb[lane + j * 32];
  }
  asm volatile("s_wait_dscnt 0" ::: "memory");
#pragma unroll 1
  for (int nt = 0; nt < 8; ++nt) {
    f32x8 acc0 = {}, acc1 = {};
#pragma unroll 1
    for (int kt = 0; kt < 12; kt += 2) {
      bf16x16 a0 = load_a_frag(A + kt * 32, UPW, lane);
      bf16x16 b0 = load_b_frag(wup, kt * 8 + nt, lane);
      bf16x16 a1 = load_a_frag(A + (kt + 1) * 32, UPW, lane);
      bf16x16 b1 = load_b_frag(wup, (kt + 1) * 8 + nt, lane);
      if (kt < 10) prefetch_b_frag(wup, (kt + 2) * 8 + nt, lane);
      acc0 = wmma_bf(a0, b0, acc0);
      acc1 = wmma_bf(a1, b1, acc1);
    }
    int n = nt * 16 + (lane & 15);
    float bias = bu[n];
    int mb = ((lane >> 4) & 1) * 8;
#pragma unroll
    for (int r = 0; r < 8; ++r) {
      size_t o = (size_t)(row0 + mb + r) * DIM + n;
      float v = h[o] + acc0[r] + acc1[r] + bias;
      h[o] = v;
      h_bf[o] = (__bf16)v;
    }
  }
}

// ---------------------------------------------------------------------------
// Split per-pair rows into padded [B,64,128] query / corpus (fp32 + bf16)
// ---------------------------------------------------------------------------
__global__ void pad_split_kernel(const float* __restrict__ h, float* __restrict__ qpad,
                                 float* __restrict__ cpad, __bf16* __restrict__ q_bf,
                                 __bf16* __restrict__ c_bf) {
  int idx = blockIdx.x * blockDim.x + threadIdx.x;  // 2 * B * 64 * 4
  int part = idx & 3;
  int row = (idx >> 2) & 63;
  int b = (idx >> 8) & 255;
  int sel = idx >> 16;
  int nvalid = sel ? NC_ : NQ_;
  int srcrow = sel ? (NQ_ + row) : row;
  float vals[32];
  if (row < nvalid) {
    const float* src = h + ((size_t)b * (NQ_ + NC_) + srcrow) * DIM + part * 32;
#pragma unroll
    for (int j = 0; j < 32; ++j) vals[j] = src[j];
  } else {
#pragma unroll
    for (int j = 0; j < 32; ++j) vals[j] = 0.0f;
  }
  size_t o = ((size_t)b * 64 + row) * DIM + part * 32;
  float* fd = (sel ? cpad : qpad) + o;
  __bf16* bd = (sel ? c_bf : q_bf) + o;
#pragma unroll
  for (int j = 0; j < 32; ++j) {
    fd[j] = vals[j];
    bd[j] = (__bf16)vals[j];
  }
}

// ---------------------------------------------------------------------------
// fc_transform: out = mask * (relu(x@w1+b1)@w2+b2); zero rows >= nvalid.
// ---------------------------------------------------------------------------
__global__ __launch_bounds__(128) void transform_kernel(
    const __bf16* __restrict__ x_bf, const __bf16* __restrict__ w1p,
    const float* __restrict__ b1, const __bf16* __restrict__ w2p,
    const float* __restrict__ b2, __bf16* __restrict__ out_bf, int nvalid) {
  __shared__ __bf16 sA[4][16 * DIM];
  __shared__ __bf16 sH[4][16 * DIM];
  int wave = threadIdx.x >> 5, lane = threadIdx.x & 31;
  int row0 = (blockIdx.x * 4 + wave) * 16;
  if (row0 >= BPAIR * 64) return;
  __bf16* A = sA[wave];
  __bf16* Hd = sH[wave];
  for (int m = 0; m < 16; ++m) {
    const unsigned* src = (const unsigned*)(x_bf + (size_t)(row0 + m) * DIM);
    unsigned* dst = (unsigned*)(A + m * DIM);
#pragma unroll
    for (int j = 0; j < 2; ++j) dst[lane + j * 32] = src[lane + j * 32];
  }
  asm volatile("s_wait_dscnt 0" ::: "memory");
#pragma unroll 1
  for (int nt = 0; nt < 8; ++nt) {
    f32x8 acc0 = {}, acc1 = {};
    {
      bf16x16 a0 = load_a_frag(A, DIM, lane);
      bf16x16 b0 = load_b_frag(w1p, nt, lane);
      bf16x16 a1 = load_a_frag(A + 32, DIM, lane);
      bf16x16 b1 = load_b_frag(w1p, 8 + nt, lane);
      acc0 = wmma_bf(a0, b0, acc0);
      acc1 = wmma_bf(a1, b1, acc1);
      a0 = load_a_frag(A + 64, DIM, lane);
      b0 = load_b_frag(w1p, 16 + nt, lane);
      a1 = load_a_frag(A + 96, DIM, lane);
      b1 = load_b_frag(w1p, 24 + nt, lane);
      acc0 = wmma_bf(a0, b0, acc0);
      acc1 = wmma_bf(a1, b1, acc1);
    }
    int n = nt * 16 + (lane & 15);
    float bias = b1[n];
    int mb = ((lane >> 4) & 1) * 8;
#pragma unroll
    for (int r = 0; r < 8; ++r) {
      float v = acc0[r] + acc1[r] + bias;
      v = v > 0.0f ? v : 0.0f;
      Hd[(mb + r) * DIM + n] = (__bf16)v;
    }
  }
#pragma unroll 1
  for (int nt = 0; nt < 8; ++nt) {
    f32x8 acc0 = {}, acc1 = {};
    {
      bf16x16 a0 = load_a_frag(Hd, DIM, lane);
      bf16x16 b0 = load_b_frag(w2p, nt, lane);
      bf16x16 a1 = load_a_frag(Hd + 32, DIM, lane);
      bf16x16 b1 = load_b_frag(w2p, 8 + nt, lane);
      acc0 = wmma_bf(a0, b0, acc0);
      acc1 = wmma_bf(a1, b1, acc1);
      a0 = load_a_frag(Hd + 64, DIM, lane);
      b0 = load_b_frag(w2p, 16 + nt, lane);
      a1 = load_a_frag(Hd + 96, DIM, lane);
      b1 = load_b_frag(w2p, 24 + nt, lane);
      acc0 = wmma_bf(a0, b0, acc0);
      acc1 = wmma_bf(a1, b1, acc1);
    }
    int n = nt * 16 + (lane & 15);
    float bias = b2[n];
    int mb = ((lane >> 4) & 1) * 8;
#pragma unroll
    for (int r = 0; r < 8; ++r) {
      int grow = row0 + mb + r;
      bool valid = (grow & 63) < nvalid;
      out_bf[(size_t)grow * DIM + n] = (__bf16)(valid ? acc0[r] + acc1[r] + bias : 0.0f);
    }
  }
}

// ---------------------------------------------------------------------------
// Per-pair fused: log_alpha = (mq@mc^T)/TEMP (WMMA), Sinkhorn x20 in LDS,
// transport = exp, score = -sum relu(qpad - T@cpad).
// ---------------------------------------------------------------------------
#define LA_ST 65
#define MC_ST 136
#define CP_ST 130
__global__ __launch_bounds__(256) void pair_kernel(
    const __bf16* __restrict__ mq_bf, const __bf16* __restrict__ mc_bf,
    const float* __restrict__ qpad, const float* __restrict__ cpad,
    float* __restrict__ out) {
  __shared__ float s_la[64 * LA_ST];
  __shared__ float s_buf[8704];  // 2 x (64 x MC_ST bf16)  OR  64 x CP_ST fp32
  __shared__ float red[256];
  int b = blockIdx.x;
  int t = threadIdx.x;
  int wave = t >> 5, lane = t & 31;
  __bf16* mqs = (__bf16*)s_buf;
  __bf16* mcs = mqs + 64 * MC_ST;
  {  // stage mq / mc tiles into LDS
    int r = t >> 2, part = t & 3;
    const unsigned* sq = (const unsigned*)(mq_bf + ((size_t)b * 64 + r) * DIM);
    const unsigned* sc = (const unsigned*)(mc_bf + ((size_t)b * 64 + r) * DIM);
    unsigned* dq = (unsigned*)(mqs + r * MC_ST);
    unsigned* dc = (unsigned*)(mcs + r * MC_ST);
#pragma unroll
    for (int j = 0; j < 16; ++j) {
      dq[part * 16 + j] = sq[part * 16 + j];
      dc[part * 16 + j] = sc[part * 16 + j];
    }
  }
  __syncthreads();
  // la = mq @ mc^T / TEMP : 16 output tiles over 8 waves
  for (int tt = wave; tt < 16; tt += 8) {
    int qt = tt >> 2, ct = tt & 3;
    f32x8 acc0 = {}, acc1 = {};
    {
      bf16x16 a0 = load_a_frag(mqs + qt * 16 * MC_ST, MC_ST, lane);
      bf16x16 b0 = load_b_frag_lds(mcs, MC_ST, ct * 16, 0, lane);
      bf16x16 a1 = load_a_frag(mqs + qt * 16 * MC_ST + 32, MC_ST, lane);
      bf16x16 b1 = load_b_frag_lds(mcs, MC_ST, ct * 16, 32, lane);
      acc0 = wmma_bf(a0, b0, acc0);
      acc1 = wmma_bf(a1, b1, acc1);
      a0 = load_a_frag(mqs + qt * 16 * MC_ST + 64, MC_ST, lane);
      b0 = load_b_frag_lds(mcs, MC_ST, ct * 16, 64, lane);
      a1 = load_a_frag(mqs + qt * 16 * MC_ST + 96, MC_ST, lane);
      b1 = load_b_frag_lds(mcs, MC_ST, ct * 16, 96, lane);
      acc0 = wmma_bf(a0, b0, acc0);
      acc1 = wmma_bf(a1, b1, acc1);
    }
    int n = ct * 16 + (lane & 15);
    int mb = ((lane >> 4) & 1) * 8;
#pragma unroll
    for (int r = 0; r < 8; ++r)
      s_la[(qt * 16 + mb + r) * LA_ST + n] = (acc0[r] + acc1[r]) * INV_TEMP;
  }
  __syncthreads();
  // Sinkhorn: 20 x (row LSE over c, col LSE over q); 4 threads per line.
  int rr = t >> 2, part = t & 3;
  for (int it = 0; it < 20; ++it) {
    float mx = -1e30f;
#pragma unroll
    for (int j = 0; j < 16; ++j) mx = fmaxf(mx, s_la[rr * LA_ST + part * 16 + j]);
    mx = fmaxf(mx, __shfl_xor(mx, 1));
    mx = fmaxf(mx, __shfl_xor(mx, 2));
    float s = 0.0f;
#pragma unroll
    for (int j = 0; j < 16; ++j) s += __expf(s_la[rr * LA_ST + part * 16 + j] - mx);
    s += __shfl_xor(s, 1);
    s += __shfl_xor(s, 2);
    float lse = mx + __logf(s);
#pragma unroll
    for (int j = 0; j < 16; ++j) s_la[rr * LA_ST + part * 16 + j] -= lse;
    __syncthreads();
    mx = -1e30f;
#pragma unroll
    for (int j = 0; j < 16; ++j) mx = fmaxf(mx, s_la[(part * 16 + j) * LA_ST + rr]);
    mx = fmaxf(mx, __shfl_xor(mx, 1));
    mx = fmaxf(mx, __shfl_xor(mx, 2));
    s = 0.0f;
#pragma unroll
    for (int j = 0; j < 16; ++j) s += __expf(s_la[(part * 16 + j) * LA_ST + rr] - mx);
    s += __shfl_xor(s, 1);
    s += __shfl_xor(s, 2);
    lse = mx + __logf(s);
#pragma unroll
    for (int j = 0; j < 16; ++j) s_la[(part * 16 + j) * LA_ST + rr] -= lse;
    __syncthreads();
  }
  // transport = exp(la); reuse s_buf for cpad tile (fp32, padded stride)
  for (int o = t; o < 64 * 64; o += 256) {
    int q = o >> 6, c = o & 63;
    s_la[q * LA_ST + c] = __expf(s_la[q * LA_ST + c]);
  }
  float* cp = s_buf;
  for (int o = t; o < 64 * DIM; o += 256) {
    int r = o >> 7, d = o & 127;
    cp[r * CP_ST + d] = cpad[((size_t)b * 64 + r) * DIM + d];
  }
  __syncthreads();
  // score = -sum relu(qpad - T @ cpad)
  float local = 0.0f;
  for (int o = t; o < 64 * DIM; o += 256) {
    int q = o >> 7, d = o & 127;
    float y = 0.0f;
    for (int c = 0; c < 64; ++c) y += s_la[q * LA_ST + c] * cp[c * CP_ST + d];
    float diff = qpad[((size_t)b * 64 + q) * DIM + d] - y;
    local += diff > 0.0f ? diff : 0.0f;
  }
  red[t] = local;
  __syncthreads();
  for (int s2 = 128; s2 > 0; s2 >>= 1) {
    if (t < s2) red[t] += red[t + s2];
    __syncthreads();
  }
  if (t == 0) out[b] = -red[0];
}

// ---------------------------------------------------------------------------
// Host launch
// ---------------------------------------------------------------------------
extern "C" void kernel_launch(void* const* d_in, const int* in_sizes, int n_in,
                              void* d_out, int out_size, void* d_ws, size_t ws_size,
                              hipStream_t stream) {
  const float* node_features = (const float*)d_in[0];
  const float* edge_features = (const float*)d_in[1];
  const int* from_idx = (const int*)d_in[2];
  const int* to_idx = (const int*)d_in[3];
  const float* enc_wn = (const float*)d_in[4];
  const float* enc_bn = (const float*)d_in[5];
  const float* enc_we = (const float*)d_in[6];
  const float* enc_be = (const float*)d_in[7];
  const float* wm1 = (const float*)d_in[8];
  const float* bm1 = (const float*)d_in[9];
  const float* wm2 = (const float*)d_in[10];
  const float* bm2 = (const float*)d_in[11];
  const float* wr1 = (const float*)d_in[12];
  const float* br1 = (const float*)d_in[13];
  const float* wr2 = (const float*)d_in[14];
  const float* br2 = (const float*)d_in[15];
  const float* wu = (const float*)d_in[16];
  const float* bu = (const float*)d_in[17];
  const float* w1 = (const float*)d_in[18];
  const float* b1 = (const float*)d_in[19];
  const float* w2 = (const float*)d_in[20];
  const float* b2 = (const float*)d_in[21];
  float* scores = (float*)d_out;

  unsigned char* wsp = (unsigned char*)d_ws;
  size_t off = 0;
  auto alloc = [&](size_t bytes) -> void* {
    void* p = wsp + off;
    off = (off + bytes + 255) & ~(size_t)255;
    return p;
  };
  float* h = (float*)alloc((size_t)N_TOTAL * DIM * 4);
  __bf16* h_bf = (__bf16*)alloc((size_t)N_TOTAL * DIM * 2);
  __bf16* e_bf = (__bf16*)alloc((size_t)E_TOTAL * ESTATE * 2);
  float* agg = (float*)alloc((size_t)N_TOTAL * MSGW * 4);
  float* qpad = (float*)alloc((size_t)BPAIR * 64 * DIM * 4);
  float* cpad = (float*)alloc((size_t)BPAIR * 64 * DIM * 4);
  __bf16* q_bf = (__bf16*)alloc((size_t)BPAIR * 64 * DIM * 2);
  __bf16* c_bf = (__bf16*)alloc((size_t)BPAIR * 64 * DIM * 2);
  __bf16* mq_bf = (__bf16*)alloc((size_t)BPAIR * 64 * DIM * 2);
  __bf16* mc_bf = (__bf16*)alloc((size_t)BPAIR * 64 * DIM * 2);
  __bf16* wnp = (__bf16*)alloc(1 * 8 * 1024);
  __bf16* wep = (__bf16*)alloc(1 * 2 * 1024);
  __bf16* wm1p = (__bf16*)alloc(9 * 16 * 1024);
  __bf16* wm2p = (__bf16*)alloc(8 * 16 * 1024);
  __bf16* wr1p = (__bf16*)alloc(9 * 16 * 1024);
  __bf16* wr2p = (__bf16*)alloc(8 * 16 * 1024);
  __bf16* wup = (__bf16*)alloc(12 * 8 * 1024);
  __bf16* w1p = (__bf16*)alloc(4 * 8 * 1024);
  __bf16* w2p = (__bf16*)alloc(4 * 8 * 1024);
  (void)ws_size; (void)n_in; (void)in_sizes; (void)out_size;

  auto pack = [&](const float* W, __bf16* outp, int K, int Nw, int kt, int nt) {
    int threads = kt * nt * 32;
    pack_weights_kernel<<<(threads + 255) / 256, 256, 0, stream>>>(W, outp, K, Nw, nt,
                                                                   threads);
  };
  pack(enc_wn, wnp, 32, 128, 1, 8);
  pack(enc_we, wep, 16, 32, 1, 2);
  pack(wm1, wm1p, 288, 256, 9, 16);
  pack(wm2, wm2p, 256, 256, 8, 16);
  pack(wr1, wr1p, 288, 256, 9, 16);
  pack(wr2, wr2p, 256, 256, 8, 16);
  pack(wu, wup, 384, 128, 12, 8);
  pack(w1, w1p, 128, 128, 4, 8);
  pack(w2, w2p, 128, 128, 4, 8);

  encode_nodes_kernel<<<N_TOTAL / 128, 256, 0, stream>>>(node_features, wnp, enc_bn, h,
                                                         h_bf);
  encode_edges_kernel<<<E_TOTAL / 128, 256, 0, stream>>>(edge_features, wep, enc_be,
                                                         e_bf);
  for (int round = 0; round < 3; ++round) {
    hipMemsetAsync(agg, 0, (size_t)N_TOTAL * MSGW * 4, stream);
    msg_pass_kernel<<<E_TOTAL / 64, 128, 0, stream>>>(h_bf, e_bf, from_idx, to_idx,
                                                      wm1p, bm1, wm2p, bm2, wr1p, br1,
                                                      wr2p, br2, agg);
    node_update_kernel<<<N_TOTAL / 64, 128, 0, stream>>>(agg, wup, bu, h, h_bf);
  }
  pad_split_kernel<<<(2 * BPAIR * 64 * 4) / 256, 256, 0, stream>>>(h, qpad, cpad, q_bf,
                                                                   c_bf);
  transform_kernel<<<(BPAIR * 64) / 64, 128, 0, stream>>>(q_bf, w1p, b1, w2p, b2, mq_bf,
                                                          NQ_);
  transform_kernel<<<(BPAIR * 64) / 64, 128, 0, stream>>>(c_bf, w1p, b1, w2p, b2, mc_bf,
                                                          NC_);
  pair_kernel<<<BPAIR, 256, 0, stream>>>(mq_bf, mc_bf, qpad, cpad, scores);
}
